// LSTM_58901181497418
// MI455X (gfx1250) — compile-verified, matrix operand order checked by reading, not run on hardware
//
#include <hip/hip_runtime.h>

// MI455X (gfx1250) LSTM forward + fused dense head, v3.
// Change vs v2: hoist all 4 A fragments of an M-tile into registers before
// the 16-WMMA block so the compiler can clause the 8 ds_load_b128 together
// and overlap the tail of the loads with the first WMMAs (partial dscnt
// waits) instead of stalling s_wait_dscnt 0x0 before every 4-WMMA group.

typedef __attribute__((ext_vector_type(16))) _Float16 v16h;
typedef __attribute__((ext_vector_type(8)))  _Float16 v8h;
typedef __attribute__((ext_vector_type(8)))  float    v8f;

#define T_STEPS 12
#define HDIM    128
#define GDIM    512             // 4*H
#define PITCH   136             // 128 + 8 pad halves -> conflict-free b128 reads
#define MROWS   64              // batch rows per workgroup
#define NMT     (MROWS / 16)    // 4 M-tiles per workgroup

// Build a v16h WMMA fragment from two contiguous 16-byte LDS chunks.
// ISA 16-bit A/B layout: element e -> K = base + e + 8*(e>=8)  (per lane-half).
__device__ __forceinline__ v16h lds_frag(const _Float16* p) {
  v8h lo = *(const v8h*)(p);        // K = base + 0..7
  v8h hi = *(const v8h*)(p + 16);   // K = base + 16..23
  v16h r;
#pragma unroll
  for (int i = 0; i < 8; ++i) { r[i] = lo[i]; r[i + 8] = hi[i]; }
  return r;
}

__device__ __forceinline__ float sigmoid_f(float z) {
  return 1.0f / (1.0f + __expf(-z));
}

__global__ __launch_bounds__(256, 1)
void lstm_wmma_kernel(const float* __restrict__ x,       // [B,T,1]
                      const float* __restrict__ kern0,   // [1,4H]
                      const float* __restrict__ rk,      // [H,4H]
                      const float* __restrict__ bias,    // [4H]
                      const float* __restrict__ Wd,      // [H,1]
                      const float* __restrict__ bd,      // [1]
                      float* __restrict__ out) {         // [B,1]
  extern __shared__ char smem[];
  _Float16* rk_s = (_Float16*)smem;   // [512][PITCH] f16 — init phase only
  _Float16* hbuf = (_Float16*)smem;   // aliased after init: 2 x [MROWS][PITCH]

  const int tid    = threadIdx.x;
  const int wave   = tid >> 5;
  const int lane   = tid & 31;
  const int laneN  = lane & 15;
  const int laneHi = lane >> 4;

  // ---- stage rkernel f32 [k][n] -> LDS f16 transposed [n][k] (one time)
  for (int idx = tid; idx < HDIM * GDIM; idx += 256) {
    int k = idx >> 9;               // 0..127
    int n = idx & 511;              // 0..511
    rk_s[n * PITCH + k] = (_Float16)rk[idx];
  }
  __syncthreads();

  // ---- persistent B fragments: this wave's 16 hidden units, all 4 gates.
  // Column n = g*128 + wave*16 + laneN ; K chunk ko*32.
  v16h bfr[4][4];
#pragma unroll
  for (int g = 0; g < 4; ++g)
#pragma unroll
    for (int ko = 0; ko < 4; ++ko)
      bfr[g][ko] = lds_frag(rk_s + (g * HDIM + wave * 16 + laneN) * PITCH +
                            ko * 32 + 8 * laneHi);
  __syncthreads();                  // everyone done reading rk_s

  // ---- h double buffers alias the staging area; h0 = 0
  for (int idx = tid; idx < 2 * MROWS * PITCH; idx += 256)
    hbuf[idx] = (_Float16)0.0f;
  __syncthreads();

  _Float16* hcur = hbuf;
  _Float16* hnxt = hbuf + MROWS * PITCH;

  const int rowbase = blockIdx.x * MROWS;
  const int j = wave * 16 + laneN;  // this lane's hidden unit (fixed)

  // Per-lane gate constants (F == 1 makes the input projection one FMA).
  const float ki  = kern0[0 * HDIM + j];
  const float kf  = kern0[1 * HDIM + j];
  const float kg  = kern0[2 * HDIM + j];
  const float ko_ = kern0[3 * HDIM + j];
  const float bi  = bias[0 * HDIM + j];
  const float bfv = bias[1 * HDIM + j];
  const float bg  = bias[2 * HDIM + j];
  const float bo  = bias[3 * HDIM + j];

  float cst[NMT][8];                // c state for this lane's (row, j) cells
#pragma unroll
  for (int a = 0; a < NMT; ++a)
#pragma unroll
    for (int b = 0; b < 8; ++b) cst[a][b] = 0.0f;

#pragma unroll 1
  for (int t = 0; t < T_STEPS; ++t) {
#pragma unroll
    for (int mt = 0; mt < NMT; ++mt) {
      // x values; acc register r maps to row m = mt*16 + r + 8*laneHi
      float xv[8];
#pragma unroll
      for (int r = 0; r < 8; ++r)
        xv[r] = x[(rowbase + mt * 16 + r + 8 * laneHi) * T_STEPS + t];

      // Hoist all 4 A fragments: 8 ds_load_b128 clause, partial waits only.
      v16h af[4];
#pragma unroll
      for (int ko = 0; ko < 4; ++ko)
        af[ko] = lds_frag(hcur + (mt * 16 + laneN) * PITCH +
                          ko * 32 + 8 * laneHi);

      v8f acc[4] = {};              // i, f, g, o accumulators
#pragma unroll
      for (int ko = 0; ko < 4; ++ko)
#pragma unroll
        for (int g = 0; g < 4; ++g)
          acc[g] = __builtin_amdgcn_wmma_f32_16x16x32_f16(
              false, af[ko], false, bfr[g][ko], (short)0, acc[g], false, false);

#pragma unroll
      for (int r = 0; r < 8; ++r) {
        float zi = acc[0][r] + xv[r] * ki  + bi;
        float zf = acc[1][r] + xv[r] * kf  + bfv;
        float zg = acc[2][r] + xv[r] * kg  + bg;
        float zo = acc[3][r] + xv[r] * ko_ + bo;
        float ig = sigmoid_f(zi);
        float fg = sigmoid_f(zf);
        float og = sigmoid_f(zo);
        float gg = fmaxf(zg, 0.0f);          // candidate activation = relu
        float c  = fg * cst[mt][r] + ig * gg;
        cst[mt][r] = c;
        float h  = og * fmaxf(c, 0.0f);      // output activation = relu
        hnxt[(mt * 16 + r + 8 * laneHi) * PITCH + j] = (_Float16)h;
      }
    }
    __syncthreads();                 // publish h(t+1) before next step reads it
    _Float16* tmp = hcur; hcur = hnxt; hnxt = tmp;
  }

  // ---- dense head: out[b] = h_T . Wd + bd. Full h_T is in hcur (LDS).
  // Wave w handles rows w*8 .. w*8+7; lane groups of 8 split the 128 columns.
  {
    const int mrow = wave * 8 + (lane & 7);
    const int jc   = (lane >> 3) * 32;       // 4 groups x 32 columns
    float s = 0.0f;
#pragma unroll
    for (int q = 0; q < 32; ++q) {
      int jj = jc + q;
      s += (float)hcur[mrow * PITCH + jj] * Wd[jj];
    }
    s += __shfl_xor(s, 8, 32);
    s += __shfl_xor(s, 16, 32);
    if ((lane >> 3) == 0)
      out[rowbase + mrow] = s + bd[0];
  }
}

extern "C" void kernel_launch(void* const* d_in, const int* in_sizes, int n_in,
                              void* d_out, int out_size, void* d_ws, size_t ws_size,
                              hipStream_t stream) {
  const float* x     = (const float*)d_in[0];
  const float* kern0 = (const float*)d_in[1];
  const float* rk    = (const float*)d_in[2];
  const float* bias  = (const float*)d_in[3];
  const float* Wd    = (const float*)d_in[4];
  const float* bd    = (const float*)d_in[5];
  float* out = (float*)d_out;

  const int B = in_sizes[0] / T_STEPS;          // F == 1
  dim3 grid((B + MROWS - 1) / MROWS);           // 1024 workgroups for B=65536
  size_t shmem = (size_t)GDIM * PITCH * 2;      // 139,264 B (h bufs alias it)
  lstm_wmma_kernel<<<grid, 256, shmem, stream>>>(x, kern0, rk, bias, Wd, bd, out);
}
